// EfficientMHSA_44461501448473
// MI455X (gfx1250) — matrix operand order repaired and checked by reading, not compiled, not run on hardware
//
#include <hip/hip_runtime.h>
#include <hip/hip_bf16.h>

// ---------------------------------------------------------------------------
// EfficientMHSA for MI455X (gfx1250): bf16 WMMA everywhere, flash-attention
// online softmax, V stored transposed so every WMMA fragment is a contiguous
// 16B/32B load.  GEMMs use double-buffered LDS tiles filled by the CDNA5
// async DMA path (global_load_async_to_lds_b128 / ASYNCcnt) so the next
// tile's transfer overlaps the current tile's 8 WMMAs.
// B=2, T=2048, C=1024, H=16, D=64, MAX_DIST=64.
// ---------------------------------------------------------------------------

typedef __bf16 bf16_t;
typedef __attribute__((ext_vector_type(16))) __bf16 v16bf;
typedef __attribute__((ext_vector_type(8)))  __bf16 v8bf;
typedef __attribute__((ext_vector_type(4)))  __bf16 v4bf;
typedef __attribute__((ext_vector_type(8)))  float  v8f;

union FragBF { v16bf v; v8bf h[2]; };

#define T_SEQ   2048
#define C_DIM   1024
#define H_HEADS 16
#define D_HEAD  64
#define LDSJ    40   // LDS row stride in halves: 32 data + 8 pad, 80B = 5*16B aligned

// ------------------------------- convert -----------------------------------
__global__ void mhsa_cvt_f32_bf16(const float* __restrict__ in,
                                  bf16_t* __restrict__ out, int n) {
    int i = (blockIdx.x * 256 + threadIdx.x) * 4;
    if (i < n) {
        float4 f = *(const float4*)(in + i);
        v4bf o;
        o[0] = (bf16_t)f.x; o[1] = (bf16_t)f.y;
        o[2] = (bf16_t)f.z; o[3] = (bf16_t)f.w;
        *(v4bf*)(out + i) = o;
    }
}

// -------- async DMA: one 32B chunk of A tile + one of B tile ---------------
// LDS dest offsets in VGPRs, 64-bit global addresses in VGPR pairs; the
// instruction offset applies to both LDS and memory sides (ISA 08 §4.4).
__device__ __forceinline__ void async_tile_load(unsigned ldsA, const bf16_t* gA,
                                                unsigned ldsB, const bf16_t* gB) {
    asm volatile(
        "global_load_async_to_lds_b128 %0, %1, off\n\t"
        "global_load_async_to_lds_b128 %0, %1, off offset:16\n\t"
        "global_load_async_to_lds_b128 %2, %3, off\n\t"
        "global_load_async_to_lds_b128 %2, %3, off offset:16"
        :: "v"(ldsA), "v"(gA), "v"(ldsB), "v"(gB) : "memory");
}

// ------------------------------- GEMM --------------------------------------
// out(M,N) = A(M,K) @ W(N,K)^T ; 128x128x32 block tiles, 8 waves (2x4),
// each wave: 64x32 = 4x2 WMMA 16x16 tiles.  Double-buffered async LDS.
// mode 0: QKV epilogue -> scatter bf16 into Q(B,H,T,D), K(B,H,T,D), Vt(B,H,D,T)
// mode 1: plain f32 store to out_f32 (row-major M x N)
__global__ __launch_bounds__(256)
void mhsa_gemm_bf16(const bf16_t* __restrict__ A, const bf16_t* __restrict__ W,
                    int M, int N, int K,
                    float* __restrict__ out_f32,
                    bf16_t* __restrict__ Qb, bf16_t* __restrict__ Kb,
                    bf16_t* __restrict__ Vt, int mode) {
    __shared__ bf16_t sA[2][128 * LDSJ];
    __shared__ bf16_t sB[2][128 * LDSJ];

    const int tid  = threadIdx.x;
    const int lane = tid & 31;
    const int wid  = tid >> 5;
    const int wm   = (wid >> 2) * 64;   // wave row base (0 / 64)
    const int wn   = (wid & 3) * 32;    // wave col base (0..96)
    const int mb   = blockIdx.x * 128;
    const int nb   = blockIdx.y * 128;

    const int lh  = lane & 15;
    const int hiL = lane >> 4;
    const int kbA = hiL * 8;            // A frag k-offset 0/8
    const int kbB = hiL * 8;            // B frag k-offset 0/8

    v8f acc[4][2];
#pragma unroll
    for (int mt = 0; mt < 4; ++mt)
#pragma unroll
        for (int nt = 0; nt < 2; ++nt)
#pragma unroll
            for (int i = 0; i < 8; ++i) acc[mt][nt][i] = 0.0f;

    const int row  = tid >> 1;          // 0..127
    const int half = tid & 1;           // which 16-half chunk of the 32-k row

    const bf16_t* gA = A + (size_t)(mb + row) * K + half * 16;
    const bf16_t* gW = W + (size_t)(nb + row) * K + half * 16;
    const unsigned ldsAoff[2] = {
        (unsigned)(uintptr_t)&sA[0][row * LDSJ + half * 16],
        (unsigned)(uintptr_t)&sA[1][row * LDSJ + half * 16] };
    const unsigned ldsBoff[2] = {
        (unsigned)(uintptr_t)&sB[0][row * LDSJ + half * 16],
        (unsigned)(uintptr_t)&sB[1][row * LDSJ + half * 16] };

    // prologue: DMA tile k0=0 into buffer 0
    async_tile_load(ldsAoff[0], gA, ldsBoff[0], gW);

    int buf = 0;
    for (int k0 = 0; k0 < K; k0 += 32) {
        asm volatile("s_wait_asynccnt 0" ::: "memory");  // my tile writes landed
        __syncthreads();                                 // everyone's landed

        FragBF af[4], bf[2];
#pragma unroll
        for (int mt = 0; mt < 4; ++mt) {
            int r = wm + mt * 16 + lh;
            af[mt].h[0] = *(const v8bf*)&sA[buf][r * LDSJ + kbA];
            af[mt].h[1] = *(const v8bf*)&sA[buf][r * LDSJ + 16 + kbA];
        }
#pragma unroll
        for (int nt = 0; nt < 2; ++nt) {
            int c = wn + nt * 16 + lh;
            bf[nt].h[0] = *(const v8bf*)&sB[buf][c * LDSJ + kbB];
            bf[nt].h[1] = *(const v8bf*)&sB[buf][c * LDSJ + kbB + 16];
        }

        // prefetch next tile into the other buffer (overlaps the WMMAs below;
        // safe: all waves' reads of buf^1 finished before this barrier round)
        if (k0 + 32 < K)
            async_tile_load(ldsAoff[buf ^ 1], gA + k0 + 32,
                            ldsBoff[buf ^ 1], gW + k0 + 32);

#pragma unroll
        for (int mt = 0; mt < 4; ++mt)
#pragma unroll
            for (int nt = 0; nt < 2; ++nt)
                acc[mt][nt] = __builtin_amdgcn_wmma_f32_16x16x32_bf16(
                    false, af[mt].v, false, bf[nt].v,
                    (short)0, acc[mt][nt], false, false);
        buf ^= 1;
    }

    const int rh = hiL * 8;             // C/D row half offset
#pragma unroll
    for (int mt = 0; mt < 4; ++mt) {
#pragma unroll
        for (int nt = 0; nt < 2; ++nt) {
#pragma unroll
            for (int i = 0; i < 8; ++i) {
                int gm = mb + wm + mt * 16 + rh + i;
                int gn = nb + wn + nt * 16 + lh;
                float v = acc[mt][nt][i];
                if (mode == 0) {
                    int b = gm >> 11, t = gm & (T_SEQ - 1);
                    int part = gn >> 10, rem = gn & (C_DIM - 1);
                    int h = rem >> 6, d = rem & (D_HEAD - 1);
                    size_t bh = (size_t)(b * H_HEADS + h);
                    if (part == 0)      Qb[(bh * T_SEQ + t) * D_HEAD + d] = (bf16_t)v;
                    else if (part == 1) Kb[(bh * T_SEQ + t) * D_HEAD + d] = (bf16_t)v;
                    else                Vt[(bh * D_HEAD + d) * T_SEQ + t] = (bf16_t)v;
                } else {
                    out_f32[(size_t)gm * N + gn] = v;
                }
            }
        }
    }
}

// --------------------------- flash attention --------------------------------
// 1 block = 64 q-rows of one (b,h); 4 waves x 16 q-rows each.
// Stream keys in chunks of 32; online softmax in WMMA C-layout registers.
__global__ __launch_bounds__(128)
void mhsa_flash_attn(const bf16_t* __restrict__ Qb, const bf16_t* __restrict__ Kb,
                     const bf16_t* __restrict__ Vt, const float* __restrict__ bias_table,
                     bf16_t* __restrict__ ctx) {
    __shared__ float  s_bias[2 * 64 + 1];        // bias_table[:, h]
    __shared__ bf16_t s_p[4 * 16 * LDSJ];        // per-wave 16x32 P patch

    const int blk  = blockIdx.x;
    const int bh   = blk >> 5;                   // 32 q-blocks per (b,h)
    const int qblk = blk & 31;
    const int h    = bh & (H_HEADS - 1);
    const int b    = bh >> 4;
    const int tid  = threadIdx.x;
    const int lane = tid & 31;
    const int wid  = tid >> 5;

    for (int i = tid; i < 129; i += 128) s_bias[i] = bias_table[i * H_HEADS + h];
    __syncthreads();

    const int lh  = lane & 15;
    const int hiL = lane >> 4;
    const int kbA = hiL * 8;
    const int kbB = hiL * 8;
    const int qbase = qblk * 64 + wid * 16;
    const int qrow0 = qbase + hiL * 8;           // element i lives on row qrow0+i

    const bf16_t* Qp = Qb + (size_t)bh * T_SEQ * D_HEAD;
    const bf16_t* Kp = Kb + (size_t)bh * T_SEQ * D_HEAD;
    const bf16_t* Vp = Vt + (size_t)bh * D_HEAD * T_SEQ;

    FragBF qf[2];
    {
        int r = qbase + lh;
#pragma unroll
        for (int f = 0; f < 2; ++f) {
            qf[f].h[0] = *(const v8bf*)&Qp[(size_t)r * D_HEAD + f * 32 + kbA];
            qf[f].h[1] = *(const v8bf*)&Qp[(size_t)r * D_HEAD + f * 32 + 16 + kbA];
        }
    }

    float m_i[8], l_i[8];
    v8f   o[4];
#pragma unroll
    for (int i = 0; i < 8; ++i) { m_i[i] = -1e30f; l_i[i] = 0.0f; }
#pragma unroll
    for (int dt = 0; dt < 4; ++dt)
#pragma unroll
        for (int i = 0; i < 8; ++i) o[dt][i] = 0.0f;

    const float scale = 0.125f;                  // D^-0.5, D=64
    bf16_t* pbuf = &s_p[wid * 16 * LDSJ];

    for (int kt0 = 0; kt0 < T_SEQ; kt0 += 32) {
        // hint the next key/value chunk toward the caches while we compute
        if (kt0 + 32 < T_SEQ) {
            __builtin_prefetch(&Kp[(size_t)(kt0 + 32 + lane) * D_HEAD], 0, 1);
            __builtin_prefetch(&Vp[(size_t)lane * T_SEQ + kt0 + 32], 0, 1);
            __builtin_prefetch(&Vp[(size_t)(lane + 32) * T_SEQ + kt0 + 32], 0, 1);
        }
        // ---- scores: two 16-key tiles, K=64 accumulated over 2 WMMA steps
        v8f s[2];
#pragma unroll
        for (int t = 0; t < 2; ++t) {
            FragBF kf[2];
            int key = kt0 + t * 16 + lh;
#pragma unroll
            for (int f = 0; f < 2; ++f) {
                kf[f].h[0] = *(const v8bf*)&Kp[(size_t)key * D_HEAD + f * 32 + kbB];
                kf[f].h[1] = *(const v8bf*)&Kp[(size_t)key * D_HEAD + f * 32 + kbB + 16];
            }
            v8f z;
#pragma unroll
            for (int i = 0; i < 8; ++i) z[i] = 0.0f;
            z = __builtin_amdgcn_wmma_f32_16x16x32_bf16(false, qf[0].v, false, kf[0].v,
                                                        (short)0, z, false, false);
            z = __builtin_amdgcn_wmma_f32_16x16x32_bf16(false, qf[1].v, false, kf[1].v,
                                                        (short)0, z, false, false);
            s[t] = z;
        }
        // ---- bias + scale + online softmax (row = lane-half-16 group)
#pragma unroll
        for (int i = 0; i < 8; ++i) {
            int qr  = qrow0 + i;
            int r0  = qr - (kt0 + lh);
            int r1  = qr - (kt0 + 16 + lh);
            r0 = r0 < -64 ? -64 : (r0 > 64 ? 64 : r0);
            r1 = r1 < -64 ? -64 : (r1 > 64 ? 64 : r1);
            float s0 = s[0][i] * scale + s_bias[r0 + 64];
            float s1 = s[1][i] * scale + s_bias[r1 + 64];
            float mx = fmaxf(s0, s1);
#pragma unroll
            for (int off = 1; off < 16; off <<= 1)
                mx = fmaxf(mx, __shfl_xor(mx, off, 32));
            float m_new = fmaxf(m_i[i], mx);
            float p0 = __expf(s0 - m_new);
            float p1 = __expf(s1 - m_new);
            float rs = p0 + p1;
#pragma unroll
            for (int off = 1; off < 16; off <<= 1)
                rs += __shfl_xor(rs, off, 32);
            float alpha = __expf(m_i[i] - m_new);
            l_i[i] = l_i[i] * alpha + rs;
            m_i[i] = m_new;
#pragma unroll
            for (int dt = 0; dt < 4; ++dt) o[dt][i] *= alpha;
            int prow = hiL * 8 + i;              // C-layout row -> LDS row
            pbuf[prow * LDSJ + lh]      = (bf16_t)p0;
            pbuf[prow * LDSJ + 16 + lh] = (bf16_t)p1;
        }
        // wave-private LDS round trip (C-layout -> A-layout); in-wave DS order
        asm volatile("s_wait_dscnt 0" ::: "memory");
        FragBF pf;
        pf.h[0] = *(const v8bf*)&pbuf[lh * LDSJ + kbA];
        pf.h[1] = *(const v8bf*)&pbuf[lh * LDSJ + 16 + kbA];
        // ---- O += P @ V  (Vt is d-major: contiguous along keys)
#pragma unroll
        for (int dt = 0; dt < 4; ++dt) {
            FragBF vf;
            int dcol = dt * 16 + lh;
            const bf16_t* vr = &Vp[(size_t)dcol * T_SEQ + kt0 + kbB];
            vf.h[0] = *(const v8bf*)&vr[0];
            vf.h[1] = *(const v8bf*)&vr[16];
            o[dt] = __builtin_amdgcn_wmma_f32_16x16x32_bf16(false, pf.v, false, vf.v,
                                                            (short)0, o[dt], false, false);
        }
    }
    // ---- finalize: O /= l, write bf16 ctx (B,T,C) with c = h*64 + d
#pragma unroll
    for (int dt = 0; dt < 4; ++dt) {
#pragma unroll
        for (int i = 0; i < 8; ++i) {
            float v = o[dt][i] / l_i[i];
            int trow = qrow0 + i;
            int c = h * D_HEAD + dt * 16 + lh;
            ctx[((size_t)b * T_SEQ + trow) * C_DIM + c] = (bf16_t)v;
        }
    }
}

// ------------------------------ launcher -----------------------------------
extern "C" void kernel_launch(void* const* d_in, const int* in_sizes, int n_in,
                              void* d_out, int out_size, void* d_ws, size_t ws_size,
                              hipStream_t stream) {
    const float* x          = (const float*)d_in[0];   // (2,2048,1024)
    const float* w_qkv      = (const float*)d_in[1];   // (3072,1024)
    const float* w_proj     = (const float*)d_in[2];   // (1024,1024)
    const float* bias_table = (const float*)d_in[3];   // (129,16)
    float* out = (float*)d_out;

    const int BT = 2 * T_SEQ;                 // 4096
    char* ws = (char*)d_ws;
    size_t off = 0;
    bf16_t* xb     = (bf16_t*)(ws + off); off += (size_t)BT * C_DIM * 2;        // 8 MB
    bf16_t* wqkvb  = (bf16_t*)(ws + off); off += (size_t)3 * C_DIM * C_DIM * 2; // 6 MB
    bf16_t* wprojb = (bf16_t*)(ws + off); off += (size_t)C_DIM * C_DIM * 2;     // 2 MB
    bf16_t* Qb     = (bf16_t*)(ws + off); off += (size_t)BT * C_DIM * 2;        // 8 MB
    bf16_t* Kb     = (bf16_t*)(ws + off); off += (size_t)BT * C_DIM * 2;        // 8 MB
    bf16_t* Vt     = (bf16_t*)(ws + off); off += (size_t)BT * C_DIM * 2;        // 8 MB
    bf16_t* ctx    = (bf16_t*)(ws + off);                                       // 8 MB

    // 1) f32 -> bf16 conversions
    {
        int n0 = BT * C_DIM, n1 = 3 * C_DIM * C_DIM, n2 = C_DIM * C_DIM;
        mhsa_cvt_f32_bf16<<<(n0 / 4 + 255) / 256, 256, 0, stream>>>(x, xb, n0);
        mhsa_cvt_f32_bf16<<<(n1 / 4 + 255) / 256, 256, 0, stream>>>(w_qkv, wqkvb, n1);
        mhsa_cvt_f32_bf16<<<(n2 / 4 + 255) / 256, 256, 0, stream>>>(w_proj, wprojb, n2);
    }
    // 2) QKV projection: (4096 x 3072 x 1024), scatter to Q/K/Vt
    mhsa_gemm_bf16<<<dim3(BT / 128, 3 * C_DIM / 128), 256, 0, stream>>>(
        xb, wqkvb, BT, 3 * C_DIM, C_DIM, nullptr, Qb, Kb, Vt, 0);
    // 3) flash attention: 2*16*(2048/64) = 1024 blocks, 128 threads
    mhsa_flash_attn<<<2 * H_HEADS * (T_SEQ / 64), 128, 0, stream>>>(
        Qb, Kb, Vt, bias_table, ctx);
    // 4) output projection: (4096 x 1024 x 1024) -> f32 d_out
    mhsa_gemm_bf16<<<dim3(BT / 128, C_DIM / 128), 256, 0, stream>>>(
        ctx, wprojb, BT, C_DIM, C_DIM, out, nullptr, nullptr, nullptr, 1);
}
